// StructureBiasedCrossAttention_82609400971465
// MI455X (gfx1250) — matrix operand order repaired and checked by reading, not compiled, not run on hardware
//
#include <hip/hip_runtime.h>
#include <hip/hip_bf16.h>

// ---------------------------------------------------------------------------
// StructureBiasedCrossAttention for MI455X (gfx1250, wave32, WMMA)
//   B=2, Q=1024, K=2048, D_MODEL=256, H=8, Dk=32, N_RBF=16, cutoff=20
// bf16 WMMA 16x16x32 with f32 accumulation; fused flash-attention with online
// softmax; RBF bias computed once per (q,k) and shared across all 8 heads via
// LDS (one block = one q-tile, 8 waves = 8 heads). Native __bf16 casts so the
// f32->bf16 conversions lower to hardware cvt ops instead of bit-twiddling.
// ---------------------------------------------------------------------------

#define DM    256
#define NH    8
#define DK    32
#define NR    16
#define QLEN  1024
#define KLEN  2048
#define BATCH 2

typedef __attribute__((ext_vector_type(16))) __bf16 v16bf;
typedef __attribute__((ext_vector_type(8)))  float  v8f;

__device__ __forceinline__ v8f wmma_bf16(v16bf a, v16bf b, v8f c) {
  return __builtin_amdgcn_wmma_f32_16x16x32_bf16(
      false, a, false, b, (short)0, c, false, false);
}

// ---------------------------------------------------------------------------
// Kernel 1: QKV projections.  job 0: Qp = qfeat@Wq^T * 1/sqrt(Dk) -> [b][h][q][d]
//                             job 1: Kp = kfeat@Wk^T              -> [b][h][k][d]
//                             job 2: Vt = (kfeat@Wv^T)^T          -> [b][h][d][k]
// One wave per 16x16 output tile; K=256 handled as 8 bf16 WMMA k-steps.
// ---------------------------------------------------------------------------
__global__ __launch_bounds__(32) void qkv_proj_kernel(
    const float* __restrict__ qfeat, const float* __restrict__ kfeat,
    const float* __restrict__ Wq, const float* __restrict__ Wk,
    const float* __restrict__ Wv,
    __bf16* __restrict__ Qp, __bf16* __restrict__ Kp, __bf16* __restrict__ Vt)
{
  const int job = blockIdx.z;
  const int mt = blockIdx.x, nt = blockIdx.y;
  const int Mrows = (job == 0) ? (BATCH * QLEN) : (BATCH * KLEN);
  if (mt * 16 >= Mrows) return;

  const float* X = (job == 0) ? qfeat : kfeat;
  const float* W = (job == 0) ? Wq : (job == 1 ? Wk : Wv);

  const int lane = threadIdx.x & 31;
  const int hlf  = lane >> 4, l16 = lane & 15;

  const float* arow = X + (size_t)(mt * 16 + l16) * DM;   // A row m = l16
  const float* brow = W + (size_t)(nt * 16 + l16) * DM;   // B col n = l16

  v8f acc = {};
  #pragma unroll
  for (int ks = 0; ks < DM; ks += 32) {
    v16bf a, b;
    #pragma unroll
    for (int e = 0; e < 8; ++e)  a[e]     = (__bf16)arow[ks + hlf * 16 + e];
    #pragma unroll
    for (int e = 0; e < 8; ++e)  a[8 + e] = (__bf16)arow[ks + hlf * 16 + 8 + e];
    #pragma unroll
    for (int e = 0; e < 16; ++e) b[e]     = (__bf16)brow[ks + hlf * 16 + e];
    acc = wmma_bf16(a, b, acc);
  }

  const float qscale = 0.17677669529663687f;   // 1/sqrt(32)
  #pragma unroll
  for (int r = 0; r < 8; ++r) {
    const int m = mt * 16 + hlf * 8 + r;       // C layout: M = r + hlf*8
    const int n = nt * 16 + l16;               //           N = l16
    const int hh = n >> 5, d = n & (DK - 1);
    const float v = acc[r];
    if (job == 0) {
      const int b = m >> 10, q = m & (QLEN - 1);
      Qp[((size_t)(b * NH + hh) * QLEN + q) * DK + d] = (__bf16)(v * qscale);
    } else if (job == 1) {
      const int b = m >> 11, k = m & (KLEN - 1);
      Kp[((size_t)(b * NH + hh) * KLEN + k) * DK + d] = (__bf16)v;
    } else {
      const int b = m >> 11, k = m & (KLEN - 1);
      Vt[((size_t)(b * NH + hh) * DK + d) * KLEN + k] = (__bf16)v;
    }
  }
}

// ---------------------------------------------------------------------------
// Kernel 2: fused biased flash-attention.
// Block = (q-tile of 16, batch); 8 waves, wave w = head w.
// Per 32-wide key tile: RBF bias computed cooperatively (once for all heads)
// into LDS; each wave does S = Q K^T (2 WMMA), bias add, online softmax,
// P transposed through wave-private LDS, O += P V (2 WMMA).
// ---------------------------------------------------------------------------
__global__ __launch_bounds__(256) void attn_kernel(
    const __bf16* __restrict__ Qp, const __bf16* __restrict__ Kp,
    const __bf16* __restrict__ Vt,
    const float* __restrict__ qpos, const float* __restrict__ kpos,
    const unsigned char* __restrict__ kmask,
    const float* __restrict__ Wrbf,
    __bf16* __restrict__ Obuf)
{
  __shared__ float  sW[NH][NR];
  __shared__ float  sQpos[16][3];
  __shared__ float  sKpos[32][3];
  __shared__ float  sKm[32];
  __shared__ float  sBias[16][32][NH];     // 16 KB
  __shared__ __bf16 sP[NH][16][32];        // 8 KB, wave-private slices

  const int tid  = threadIdx.x;
  const int qt   = blockIdx.x;             // 0..63
  const int b    = blockIdx.y;             // 0..1
  const int h    = tid >> 5;
  const int lane = tid & 31, hlf = lane >> 4, l16 = lane & 15;

  if (tid < NH * NR) sW[tid >> 4][tid & 15] = Wrbf[tid];
  if (tid < 48) sQpos[tid / 3][tid % 3] = qpos[(size_t)(b * QLEN + qt * 16) * 3 + tid];

  // Q fragment for this head (constant over the whole k loop)
  v16bf qf;
  {
    const __bf16* qr = Qp + ((size_t)(b * NH + h) * QLEN + qt * 16 + l16) * DK;
    #pragma unroll
    for (int e = 0; e < 8; ++e) qf[e]     = qr[hlf * 16 + e];
    #pragma unroll
    for (int e = 0; e < 8; ++e) qf[8 + e] = qr[hlf * 16 + 8 + e];
  }

  float rowM[8], rowL[8];
  v8f o0 = {}, o1 = {};
  #pragma unroll
  for (int r = 0; r < 8; ++r) { rowM[r] = -3.0e38f; rowL[r] = 0.0f; }

  const __bf16* kbase = Kp + (size_t)(b * NH + h) * KLEN * DK;
  const __bf16* vbase = Vt + (size_t)(b * NH + h) * DK * KLEN;

  for (int kt = 0; kt < KLEN / 32; ++kt) {
    __syncthreads();   // protect sKpos/sBias reuse across iterations
    if (tid < 96) sKpos[tid / 3][tid % 3] = kpos[(size_t)(b * KLEN + kt * 32) * 3 + tid];
    if (tid < 32) sKm[tid] = kmask[(size_t)b * KLEN + kt * 32 + tid] ? 0.0f : -1.0e30f;
    __syncthreads();

    // --- cooperative RBF bias: 512 (q,k) pairs, 2 per thread, all heads ---
    #pragma unroll
    for (int pp = 0; pp < 2; ++pp) {
      const int p  = tid * 2 + pp;
      const int ql = p >> 5, kl = p & 31;
      const float dx = sQpos[ql][0] - sKpos[kl][0];
      const float dy = sQpos[ql][1] - sKpos[kl][1];
      const float dz = sQpos[ql][2] - sKpos[kl][2];
      const float d  = sqrtf(dx * dx + dy * dy + dz * dz);
      float acc[NH];
      #pragma unroll
      for (int hh = 0; hh < NH; ++hh) acc[hh] = sKm[kl];
      #pragma unroll
      for (int r = 0; r < NR; ++r) {                    // centres = r*20/15
        const float t = d - (float)r * (20.0f / 15.0f);
        const float e = __expf(t * t * -0.32f);         // 1/(2*1.25^2)=0.32
        #pragma unroll
        for (int hh = 0; hh < NH; ++hh) acc[hh] = fmaf(e, sW[hh][r], acc[hh]);
      }
      #pragma unroll
      for (int hh = 0; hh < NH; ++hh) sBias[ql][kl][hh] = acc[hh];
    }
    __syncthreads();

    // --- S = Q K^T for this head (two 16-col halves of the 32-key tile) ---
    v16bf kf0, kf1;
    {
      const __bf16* kr0 = kbase + (size_t)(kt * 32 + l16) * DK + hlf * 16;
      const __bf16* kr1 = kr0 + 16 * DK;
      #pragma unroll
      for (int e = 0; e < 16; ++e) kf0[e] = kr0[e];
      #pragma unroll
      for (int e = 0; e < 16; ++e) kf1[e] = kr1[e];
    }
    if (kt + 1 < KLEN / 32) {   // prefetch next K/V tiles (global_prefetch_b8)
      __builtin_prefetch(kbase + (size_t)(kt * 32 + 32 + l16) * DK, 0, 3);
      __builtin_prefetch(vbase + (size_t)l16 * KLEN + kt * 32 + 32, 0, 3);
    }
    v8f s0 = wmma_bf16(qf, kf0, (v8f){});
    v8f s1 = wmma_bf16(qf, kf1, (v8f){});
    #pragma unroll
    for (int r = 0; r < 8; ++r) {
      s0[r] += sBias[hlf * 8 + r][l16][h];
      s1[r] += sBias[hlf * 8 + r][16 + l16][h];
    }

    // --- online softmax (row m = hlf*8 + r lives across the 16-lane half) ---
    float corr[8];
    #pragma unroll
    for (int r = 0; r < 8; ++r) {
      float mx = fmaxf(s0[r], s1[r]);
      #pragma unroll
      for (int msk = 1; msk < 16; msk <<= 1) mx = fmaxf(mx, __shfl_xor(mx, msk, 32));
      const float nm = fmaxf(rowM[r], mx);
      corr[r] = __expf(rowM[r] - nm);
      rowM[r] = nm;
      const float p0 = __expf(s0[r] - nm);
      const float p1 = __expf(s1[r] - nm);
      s0[r] = p0; s1[r] = p1;
      float sum = p0 + p1;
      #pragma unroll
      for (int msk = 1; msk < 16; msk <<= 1) sum += __shfl_xor(sum, msk, 32);
      rowL[r] = rowL[r] * corr[r] + sum;
    }

    // --- P: C-layout -> A-layout via wave-private LDS tile ---
    #pragma unroll
    for (int r = 0; r < 8; ++r) {
      sP[h][hlf * 8 + r][l16]      = (__bf16)s0[r];
      sP[h][hlf * 8 + r][16 + l16] = (__bf16)s1[r];
    }
    v16bf pf;
    {
      const __bf16* pr = &sP[h][l16][0];
      #pragma unroll
      for (int e = 0; e < 8; ++e) pf[e]     = pr[hlf * 16 + e];
      #pragma unroll
      for (int e = 0; e < 8; ++e) pf[8 + e] = pr[hlf * 16 + 8 + e];
    }

    // --- O += P V   (V^T layout makes B-frags contiguous per lane) ---
    v16bf vf0, vf1;
    {
      const __bf16* vr0 = vbase + (size_t)l16 * KLEN + kt * 32 + hlf * 16;
      const __bf16* vr1 = vbase + (size_t)(16 + l16) * KLEN + kt * 32 + hlf * 16;
      #pragma unroll
      for (int e = 0; e < 16; ++e) vf0[e] = vr0[e];
      #pragma unroll
      for (int e = 0; e < 16; ++e) vf1[e] = vr1[e];
    }
    #pragma unroll
    for (int r = 0; r < 8; ++r) { o0[r] *= corr[r]; o1[r] *= corr[r]; }
    o0 = wmma_bf16(pf, vf0, o0);
    o1 = wmma_bf16(pf, vf1, o1);
  }

  // epilogue: O / L -> Obuf[b][q][h*32+d] as bf16 for the Wo GEMM
  #pragma unroll
  for (int r = 0; r < 8; ++r) {
    const int m = hlf * 8 + r;
    const size_t row = (size_t)b * QLEN + qt * 16 + m;
    const float inv = 1.0f / rowL[r];
    Obuf[row * DM + h * DK + l16]      = (__bf16)(o0[r] * inv);
    Obuf[row * DM + h * DK + 16 + l16] = (__bf16)(o1[r] * inv);
  }
}

// ---------------------------------------------------------------------------
// Kernel 3: Y = Obuf @ Wo^T + bo ; X = query_feat + Y*query_mask  (pre-LN)
// ---------------------------------------------------------------------------
__global__ __launch_bounds__(32) void oproj_kernel(
    const __bf16* __restrict__ Obuf, const float* __restrict__ Wo,
    const float* __restrict__ bo, const float* __restrict__ qfeat,
    const unsigned char* __restrict__ qmask, float* __restrict__ X)
{
  const int mt = blockIdx.x, nt = blockIdx.y;
  const int lane = threadIdx.x & 31, hlf = lane >> 4, l16 = lane & 15;

  const __bf16* arow = Obuf + (size_t)(mt * 16 + l16) * DM;
  const float*  brow = Wo   + (size_t)(nt * 16 + l16) * DM;

  v8f acc = {};
  #pragma unroll
  for (int ks = 0; ks < DM; ks += 32) {
    v16bf a, bfr;
    #pragma unroll
    for (int e = 0; e < 8; ++e)  a[e]     = arow[ks + hlf * 16 + e];
    #pragma unroll
    for (int e = 0; e < 8; ++e)  a[8 + e] = arow[ks + hlf * 16 + 8 + e];
    #pragma unroll
    for (int e = 0; e < 16; ++e) bfr[e]   = (__bf16)brow[ks + hlf * 16 + e];
    acc = wmma_bf16(a, bfr, acc);
  }

  #pragma unroll
  for (int r = 0; r < 8; ++r) {
    const int m = mt * 16 + hlf * 8 + r;
    const int n = nt * 16 + l16;
    const float y  = acc[r] + bo[n];
    const float xv = qfeat[(size_t)m * DM + n] + (qmask[m] ? y : 0.0f);
    X[(size_t)m * DM + n] = xv;
  }
}

// ---------------------------------------------------------------------------
// Kernel 4: LayerNorm, one wave per 256-element row (8 values per lane).
// ---------------------------------------------------------------------------
__global__ __launch_bounds__(256) void ln_kernel(
    const float* __restrict__ X, const float* __restrict__ g,
    const float* __restrict__ bta, float* __restrict__ out)
{
  const int wave = threadIdx.x >> 5, lane = threadIdx.x & 31;
  const size_t row = (size_t)blockIdx.x * 8 + wave;   // 2048 rows total
  const float* xr = X + row * DM + lane * 8;

  float v[8];
  #pragma unroll
  for (int i = 0; i < 8; ++i) v[i] = xr[i];

  float s = 0.0f;
  #pragma unroll
  for (int i = 0; i < 8; ++i) s += v[i];
  #pragma unroll
  for (int msk = 1; msk < 32; msk <<= 1) s += __shfl_xor(s, msk, 32);
  const float mu = s * (1.0f / 256.0f);

  float vs = 0.0f;
  #pragma unroll
  for (int i = 0; i < 8; ++i) { const float d = v[i] - mu; vs += d * d; }
  #pragma unroll
  for (int msk = 1; msk < 32; msk <<= 1) vs += __shfl_xor(vs, msk, 32);
  const float rstd = rsqrtf(vs * (1.0f / 256.0f) + 1e-5f);

  #pragma unroll
  for (int i = 0; i < 8; ++i) {
    const int c = lane * 8 + i;
    out[row * DM + c] = (v[i] - mu) * rstd * g[c] + bta[c];
  }
}

// ---------------------------------------------------------------------------
extern "C" void kernel_launch(void* const* d_in, const int* in_sizes, int n_in,
                              void* d_out, int out_size, void* d_ws, size_t ws_size,
                              hipStream_t stream)
{
  const float*         qfeat = (const float*)d_in[0];
  const float*         kfeat = (const float*)d_in[1];
  const float*         qpos  = (const float*)d_in[2];
  const float*         kpos  = (const float*)d_in[3];
  const unsigned char* qmask = (const unsigned char*)d_in[4];
  const unsigned char* kmask = (const unsigned char*)d_in[5];
  const float*         Wq    = (const float*)d_in[6];
  const float*         Wk    = (const float*)d_in[7];
  const float*         Wv    = (const float*)d_in[8];
  const float*         Wo    = (const float*)d_in[9];
  const float*         bo    = (const float*)d_in[10];
  const float*         Wrbf  = (const float*)d_in[11];
  const float*         ln_g  = (const float*)d_in[12];
  const float*         ln_b  = (const float*)d_in[13];

  char* ws = (char*)d_ws;                        // ~8 MB used
  __bf16* Qp = (__bf16*)(ws);                    // 1 MB  [b][h][q][d]
  __bf16* Kp = (__bf16*)(ws + (1 << 20));        // 2 MB  [b][h][k][d]
  __bf16* Vt = (__bf16*)(ws + (3 << 20));        // 2 MB  [b][h][d][k]
  __bf16* Ob = (__bf16*)(ws + (5 << 20));        // 1 MB  [b][q][h*32+d]
  float*  X  = (float*)(ws + (6 << 20));         // 2 MB  pre-LN rows

  dim3 g1(256, 16, 3);                    // m-tiles (max), n-tiles, {Q,K,V}
  qkv_proj_kernel<<<g1, 32, 0, stream>>>(qfeat, kfeat, Wq, Wk, Wv, Qp, Kp, Vt);

  dim3 g2(QLEN / 16, BATCH);              // 64 q-tiles x 2 batches, 8 waves/blk
  attn_kernel<<<g2, 256, 0, stream>>>(Qp, Kp, Vt, qpos, kpos, kmask, Wrbf, Ob);

  dim3 g3((BATCH * QLEN) / 16, DM / 16);  // 128 x 16 tiles
  oproj_kernel<<<g3, 32, 0, stream>>>(Ob, Wo, bo, qfeat, qmask, X);

  ln_kernel<<<(BATCH * QLEN) / 8, 256, 0, stream>>>(X, ln_g, ln_b, (float*)d_out);
}
